// LSTMActorCritic_23132693856668
// MI455X (gfx1250) — compile-verified, matrix operand order checked by reading, not run on hardware
//
#include <hip/hip_runtime.h>
#include <hip/hip_bf16.h>

// ---------------------------------------------------------------------------
// Types
// ---------------------------------------------------------------------------
typedef __attribute__((ext_vector_type(16))) __bf16     bf16x16;
typedef __attribute__((ext_vector_type(8)))  float      f32x8;
typedef __attribute__((ext_vector_type(4)))  unsigned   u32x4;
typedef __attribute__((ext_vector_type(8)))  int        i32x8;
typedef __attribute__((ext_vector_type(4)))  int        i32x4;

#define T_STEPS 256
#define BATCH   256
#define OBS     128
#define HID     256
#define HSZ     512
#define ACT     16

#define MU_OFF   0
#define SIG_OFF  (T_STEPS * BATCH * ACT)          // 1048576
#define VAL_OFF  (2 * T_STEPS * BATCH * ACT)      // 2097152

// ---------------------------------------------------------------------------
// Fragment loaders (layouts per CDNA5 ISA §7.12.2)
//   A (16x32 bf16): lane m<16 -> row m, K = kt*32 + {0..7, 16..23}
//                   lane m>=16 -> row m-16, K = kt*32 + {8..15, 24..31}
//   B packed offline so each lane reads 16 contiguous bf16.
// ---------------------------------------------------------------------------
__device__ __forceinline__ bf16x16 load_b_frag(const __bf16* __restrict__ packed,
                                               int frag, int lane) {
  const u32x4* p = (const u32x4*)(packed + (size_t)frag * 512) + lane * 2;
  union { bf16x16 v; u32x4 q[2]; } u;
  u.q[0] = p[0];
  u.q[1] = p[1];
  return u.v;
}

__device__ __forceinline__ bf16x16 load_a_frag_bf16(const __bf16* base, int ld,
                                                    int kt, int lane) {
  int row   = lane & 15;
  int kbase = kt * 32 + ((lane >> 4) << 3);
  const __bf16* p = base + (size_t)row * ld + kbase;
  union { bf16x16 v; u32x4 q[2]; } u;
  u.q[0] = *(const u32x4*)p;
  u.q[1] = *(const u32x4*)(p + 16);
  return u.v;
}

__device__ __forceinline__ bf16x16 load_a_frag_f32(const float* base, int ld,
                                                   int kt, int lane) {
  int row   = lane & 15;
  int kbase = kt * 32 + ((lane >> 4) << 3);
  const float* p = base + (size_t)row * ld + kbase;
  bf16x16 v;
#pragma unroll
  for (int j = 0; j < 8; ++j) {
    v[j]     = (__bf16)p[j];
    v[8 + j] = (__bf16)p[16 + j];
  }
  return v;
}

__device__ __forceinline__ f32x8 wmma_bf16(bf16x16 a, bf16x16 b, f32x8 c) {
  return __builtin_amdgcn_wmma_f32_16x16x32_bf16(false, a, false, b,
                                                 (short)0, c, false, false);
}

__device__ __forceinline__ float sigmoidf_(float x) {
  return 1.0f / (1.0f + __expf(-x));
}
__device__ __forceinline__ float gelu_tanh(float x) {
  float x3 = x * x * x;
  return 0.5f * x * (1.0f + tanhf(0.7978845608028654f * (x + 0.044715f * x3)));
}

// ---------------------------------------------------------------------------
// TDM: async 2D tile load Global -> LDS (Tensor Data Mover).
// Descriptor per CDNA5 ISA (D# groups 0/1; groups 2/3
// zero => tensors up to 2D). data_size = 1 byte, dims in bytes.
// Tracked with TENSORcnt; wait via s_wait_tensorcnt.
// NOTE: this toolchain exposes the 6-argument builtin form
//   (uint32x4 g0, int32x8 g1, int32x4, int32x4, int32x8, i32 cpol)
// ---------------------------------------------------------------------------
__device__ __forceinline__ unsigned lds_byte_addr(const void* p) {
  return (unsigned)(unsigned long long)
      (__attribute__((address_space(3))) const void*)p;
}

__device__ __forceinline__ void tdm_load_tile_2d(unsigned lds_addr,
                                                 const void* gptr,
                                                 unsigned row_bytes,
                                                 unsigned nrows,
                                                 unsigned stride_bytes) {
  unsigned long long ga = (unsigned long long)gptr;
  u32x4 g0;
  g0[0] = 1u;                                              // count=1 (valid D#)
  g0[1] = lds_addr;                                        // lds_addr[31:0]
  g0[2] = (unsigned)(ga & 0xFFFFFFFFu);                    // global_addr[31:0]
  g0[3] = (unsigned)((ga >> 32) & 0x01FFFFFFu) | 0x80000000u; // ga[56:32]|type=2
  i32x8 g1;
  g1[0] = 0;                                               // wg_mask=0, dsize=1B
  g1[1] = (int)((row_bytes & 0xFFFFu) << 16);              // tensor_dim0[15:0]
  g1[2] = (int)((row_bytes >> 16) | ((nrows & 0xFFFFu) << 16)); // d0 hi | d1 lo
  g1[3] = (int)(((nrows >> 16) & 0xFFFFu) | ((row_bytes & 0xFFFFu) << 16)); // d1 hi | tile_dim0
  g1[4] = (int)(nrows & 0xFFFFu);                          // tile_dim1 | tile_dim2=0
  g1[5] = (int)stride_bytes;                               // dim0_stride[31:0]
  g1[6] = 0;                                               // stride hi | d1_stride lo
  g1[7] = 0;
  i32x4 z4 = {0, 0, 0, 0};
  i32x8 z8 = {0, 0, 0, 0, 0, 0, 0, 0};
  __builtin_amdgcn_tensor_load_to_lds(g0, g1, z4, z4, z8, 0);
}

// ---------------------------------------------------------------------------
// Kernel 1: pack a KxN fp32 row-major matrix into bf16 WMMA B-fragments.
// One wave per 32x16 fragment; lane l writes 16 contiguous bf16.
// frag id = ntile*(K/32) + kt.  N may be padded (cols >= Nsrc become zero).
// ---------------------------------------------------------------------------
__global__ __launch_bounds__(32) void pack_b_kernel(
    const float* __restrict__ src, __bf16* __restrict__ dst,
    int KT, int Nsrc, int ldsrc) {
  int fi    = blockIdx.x;
  int lane  = threadIdx.x;
  int ntile = fi / KT;
  int kt    = fi - ntile * KT;
  int col   = ntile * 16 + (lane & 15);
  int kbase = kt * 32 + ((lane >> 4) << 3);

  bf16x16 v;
#pragma unroll
  for (int j = 0; j < 8; ++j) {
    float a = (col < Nsrc) ? src[(size_t)(kbase + j) * ldsrc + col]      : 0.0f;
    float b = (col < Nsrc) ? src[(size_t)(kbase + 16 + j) * ldsrc + col] : 0.0f;
    v[j]     = (__bf16)a;
    v[8 + j] = (__bf16)b;
  }
  union { bf16x16 vv; u32x4 q[2]; } u;
  u.vv = v;
  u32x4* out = (u32x4*)(dst + (size_t)fi * 512) + lane * 2;
  out[0] = u.q[0];
  out[1] = u.q[1];
}

// ---------------------------------------------------------------------------
// Kernel 2: feats = gelu(x @ W_enc + b_enc), bf16 row-major (65536 x 256).
// One WG = 16 rows; 8 waves x 2 column-tiles; K=128 -> 4 ktiles.
// ---------------------------------------------------------------------------
__global__ __launch_bounds__(256) void encoder_kernel(
    const float* __restrict__ x, const __bf16* __restrict__ wencP,
    const float* __restrict__ b_enc, __bf16* __restrict__ feats) {
  int row0 = blockIdx.x * 16;
  int lane = threadIdx.x & 31;
  int w    = threadIdx.x >> 5;
  const float* xbase = x + (size_t)row0 * OBS;

#pragma unroll
  for (int s = 0; s < 2; ++s) {
    int nt = w * 2 + s;
    f32x8 acc;
    float bv = b_enc[nt * 16 + (lane & 15)];
#pragma unroll
    for (int r = 0; r < 8; ++r) acc[r] = bv;

    for (int kt = 0; kt < 4; ++kt) {
      bf16x16 a = load_a_frag_f32(xbase, OBS, kt, lane);
      bf16x16 b = load_b_frag(wencP, nt * 4 + kt, lane);
      acc = wmma_bf16(a, b, acc);
    }
    int col = nt * 16 + (lane & 15);
#pragma unroll
    for (int r = 0; r < 8; ++r) {
      int m = r + ((lane >> 4) << 3);
      feats[(size_t)(row0 + m) * HID + col] = (__bf16)gelu_tanh(acc[r]);
    }
  }
}

// ---------------------------------------------------------------------------
// Kernel 3: persistent LSTM scan.
// Grid = 16 WGs; each owns 16 batch rows for all 256 timesteps (batch rows
// are independent through the recurrence -> no inter-WG sync).
//  - h state: double-buffered bf16 in LDS (32 KB); c state in registers.
//  - feats tile (16x256 bf16 = 8 KB) double-buffered in LDS, prefetched one
//    step ahead by wave 0 via TDM tensor_load_to_lds (TENSORcnt).
//  - h_t also streamed to global `hs` for the separate heads kernel.
// Per wave per step: 4 col-tiles x 4 gates x (8 Wi + 16 Wh) ktiles = 384 WMMA.
// ---------------------------------------------------------------------------
__global__ __launch_bounds__(256) void lstm_kernel(
    const __bf16* __restrict__ feats, const __bf16* __restrict__ wiP,
    const __bf16* __restrict__ whP, const float* __restrict__ b_lstm,
    __bf16* __restrict__ hs) {
  __shared__ __bf16 h_lds[2][16 * HSZ];   // 32 KB
  __shared__ __bf16 f_lds[2][16 * HID];   // 16 KB

  int lane = threadIdx.x & 31;
  int w    = threadIdx.x >> 5;
  int b0   = blockIdx.x * 16;

  // zero-init both h buffers
  __bf16* hflat = &h_lds[0][0];
  for (int i = threadIdx.x; i < 2 * 16 * HSZ; i += 256) hflat[i] = (__bf16)0.0f;

  // prefetch feats tile for t=0 (TDM, wave 0 only; TENSOR ops ignore EXEC so
  // issue from a single wave-uniform branch)
  if (w == 0) {
    tdm_load_tile_2d(lds_byte_addr(&f_lds[0][0]),
                     feats + (size_t)b0 * HID,
                     HID * 2, 16, HID * 2);
    __builtin_amdgcn_s_wait_tensorcnt(0);
  }
  __syncthreads();

  f32x8 c_reg[4];
#pragma unroll
  for (int ct = 0; ct < 4; ++ct)
#pragma unroll
    for (int r = 0; r < 8; ++r) c_reg[ct][r] = 0.0f;

  float bias[4][4];
#pragma unroll
  for (int ct = 0; ct < 4; ++ct)
#pragma unroll
    for (int g = 0; g < 4; ++g)
      bias[ct][g] = b_lstm[g * HSZ + w * 64 + ct * 16 + (lane & 15)];

  int cur = 0;
  for (int t = 0; t < T_STEPS; ++t) {
    const __bf16* hcur  = &h_lds[cur][0];
    __bf16*       hnxt  = &h_lds[cur ^ 1][0];
    const __bf16* fbase = &f_lds[t & 1][0];
    __bf16*       hsrow = hs + (size_t)(t * BATCH + b0) * HSZ;

    // kick off async prefetch of next step's feats tile while we compute
    if (w == 0 && (t + 1) < T_STEPS) {
      tdm_load_tile_2d(lds_byte_addr(&f_lds[(t + 1) & 1][0]),
                       feats + (size_t)((t + 1) * BATCH + b0) * HID,
                       HID * 2, 16, HID * 2);
    }

#pragma unroll
    for (int ct = 0; ct < 4; ++ct) {
      int col0 = w * 64 + ct * 16;
      f32x8 acc0, acc1, acc2, acc3;
#pragma unroll
      for (int r = 0; r < 8; ++r) {
        acc0[r] = bias[ct][0];
        acc1[r] = bias[ct][1];
        acc2[r] = bias[ct][2];
        acc3[r] = bias[ct][3];
      }
      int nt_i = (0 * HSZ + col0) >> 4;
      int nt_f = (1 * HSZ + col0) >> 4;
      int nt_g = (2 * HSZ + col0) >> 4;
      int nt_o = (3 * HSZ + col0) >> 4;

      // input-gate contribution: z += feats_t @ Wi   (K = 256, A from LDS)
      for (int kt = 0; kt < 8; ++kt) {
        bf16x16 a = load_a_frag_bf16(fbase, HID, kt, lane);
        acc0 = wmma_bf16(a, load_b_frag(wiP, nt_i * 8 + kt, lane), acc0);
        acc1 = wmma_bf16(a, load_b_frag(wiP, nt_f * 8 + kt, lane), acc1);
        acc2 = wmma_bf16(a, load_b_frag(wiP, nt_g * 8 + kt, lane), acc2);
        acc3 = wmma_bf16(a, load_b_frag(wiP, nt_o * 8 + kt, lane), acc3);
      }
      // recurrent contribution: z += h @ Wh          (K = 512, A from LDS)
      for (int kt = 0; kt < 16; ++kt) {
        bf16x16 a = load_a_frag_bf16(hcur, HSZ, kt, lane);
        acc0 = wmma_bf16(a, load_b_frag(whP, nt_i * 16 + kt, lane), acc0);
        acc1 = wmma_bf16(a, load_b_frag(whP, nt_f * 16 + kt, lane), acc1);
        acc2 = wmma_bf16(a, load_b_frag(whP, nt_g * 16 + kt, lane), acc2);
        acc3 = wmma_bf16(a, load_b_frag(whP, nt_o * 16 + kt, lane), acc3);
      }
      // gate nonlinearities; publish h_new to LDS (recurrence) + global (heads)
      int col = col0 + (lane & 15);
#pragma unroll
      for (int r = 0; r < 8; ++r) {
        float iv = sigmoidf_(acc0[r]);
        float fv = sigmoidf_(acc1[r]);
        float gv = tanhf(acc2[r]);
        float ov = sigmoidf_(acc3[r]);
        float cn = fv * c_reg[ct][r] + iv * gv;
        float hn = ov * tanhf(cn);
        c_reg[ct][r] = cn;
        int m = r + ((lane >> 4) << 3);
        __bf16 hb = (__bf16)hn;
        hnxt[m * HSZ + col]  = hb;
        hsrow[m * HSZ + col] = hb;
      }
    }

    // make next feats tile visible before anyone reads it next step
    if (w == 0 && (t + 1) < T_STEPS) __builtin_amdgcn_s_wait_tensorcnt(0);
    __syncthreads();
    cur ^= 1;
  }
}

// ---------------------------------------------------------------------------
// Kernel 4: heads over all (t,b) rows, full-grid parallel.
//   mu = hs @ W_mu + b_mu ; value = hs @ W_v + b_v (W_v zero-padded to 16 cols)
// One WG = 16 rows, 2 waves (wave0: mu, wave1: value), K=512 -> 16 ktiles.
// ---------------------------------------------------------------------------
__global__ __launch_bounds__(64) void head_kernel(
    const __bf16* __restrict__ hs, const __bf16* __restrict__ wmuP,
    const __bf16* __restrict__ wvP, const float* __restrict__ b_mu,
    const float* __restrict__ b_v, float* __restrict__ out) {
  int row0 = blockIdx.x * 16;
  int lane = threadIdx.x & 31;
  int w    = threadIdx.x >> 5;
  const __bf16* hbase = hs + (size_t)row0 * HSZ;
  const __bf16* bP    = (w == 0) ? wmuP : wvP;
  float bv = (w == 0) ? b_mu[lane & 15] : b_v[0];

  f32x8 acc;
#pragma unroll
  for (int r = 0; r < 8; ++r) acc[r] = bv;
  for (int kt = 0; kt < 16; ++kt) {
    bf16x16 a = load_a_frag_bf16(hbase, HSZ, kt, lane);
    acc = wmma_bf16(a, load_b_frag(bP, kt, lane), acc);
  }
  if (w == 0) {
#pragma unroll
    for (int r = 0; r < 8; ++r) {
      int m = r + ((lane >> 4) << 3);
      out[MU_OFF + (size_t)(row0 + m) * ACT + (lane & 15)] = acc[r];
    }
  } else if ((lane & 15) == 0) {
#pragma unroll
    for (int r = 0; r < 8; ++r) {
      int m = r + ((lane >> 4) << 3);
      out[VAL_OFF + (size_t)(row0 + m)] = acc[r];
    }
  }
}

// ---------------------------------------------------------------------------
// Kernel 5: sigma = broadcast(exp(log_std))
// ---------------------------------------------------------------------------
__global__ __launch_bounds__(256) void sigma_kernel(
    const float* __restrict__ log_std, float* __restrict__ out_sigma) {
  int i = blockIdx.x * blockDim.x + threadIdx.x;
  out_sigma[i] = __expf(log_std[i & (ACT - 1)]);
}

// ---------------------------------------------------------------------------
// Launch
// ---------------------------------------------------------------------------
extern "C" void kernel_launch(void* const* d_in, const int* in_sizes, int n_in,
                              void* d_out, int out_size, void* d_ws,
                              size_t ws_size, hipStream_t stream) {
  const float* x       = (const float*)d_in[0];
  const float* W_enc   = (const float*)d_in[1];
  const float* b_enc   = (const float*)d_in[2];
  const float* Wi      = (const float*)d_in[3];
  const float* Wh      = (const float*)d_in[4];
  const float* b_lstm  = (const float*)d_in[5];
  const float* W_mu    = (const float*)d_in[6];
  const float* b_mu    = (const float*)d_in[7];
  const float* log_std = (const float*)d_in[8];
  const float* W_v     = (const float*)d_in[9];
  const float* b_v     = (const float*)d_in[10];
  float* out = (float*)d_out;
  char*  ws  = (char*)d_ws;

  // workspace layout
  size_t off = 0;
  __bf16* feats = (__bf16*)(ws + off); off += (size_t)T_STEPS * BATCH * HID * 2; // 32 MB
  __bf16* hsbuf = (__bf16*)(ws + off); off += (size_t)T_STEPS * BATCH * HSZ * 2; // 64 MB
  __bf16* wencP = (__bf16*)(ws + off); off += 64 * 1024;
  __bf16* wiP   = (__bf16*)(ws + off); off += 1024 * 1024;
  __bf16* whP   = (__bf16*)(ws + off); off += 2048 * 1024;
  __bf16* wmuP  = (__bf16*)(ws + off); off += 16 * 1024;
  __bf16* wvP   = (__bf16*)(ws + off); off += 16 * 1024;

  // 1) pack weights into WMMA B-fragment layout (bf16)
  pack_b_kernel<<<64,   32, 0, stream>>>(W_enc, wencP, /*KT=*/4,  /*Nsrc=*/256,  /*ld=*/256);
  pack_b_kernel<<<1024, 32, 0, stream>>>(Wi,    wiP,   /*KT=*/8,  /*Nsrc=*/2048, /*ld=*/2048);
  pack_b_kernel<<<2048, 32, 0, stream>>>(Wh,    whP,   /*KT=*/16, /*Nsrc=*/2048, /*ld=*/2048);
  pack_b_kernel<<<16,   32, 0, stream>>>(W_mu,  wmuP,  /*KT=*/16, /*Nsrc=*/16,   /*ld=*/16);
  pack_b_kernel<<<16,   32, 0, stream>>>(W_v,   wvP,   /*KT=*/16, /*Nsrc=*/1,    /*ld=*/1);

  // 2) encoder GEMM + gelu -> feats (bf16)
  encoder_kernel<<<(T_STEPS * BATCH) / 16, 256, 0, stream>>>(x, wencP, b_enc, feats);

  // 3) fused LSTM scan (input GEMM + recurrence), TDM-prefetched feats tiles
  lstm_kernel<<<BATCH / 16, 256, 0, stream>>>(feats, wiP, whP, b_lstm, hsbuf);

  // 4) heads over all rows (full-machine parallel)
  head_kernel<<<(T_STEPS * BATCH) / 16, 64, 0, stream>>>(hsbuf, wmuP, wvP,
                                                         b_mu, b_v, out);

  // 5) sigma broadcast
  sigma_kernel<<<(T_STEPS * BATCH * ACT) / 256, 256, 0, stream>>>(log_std,
                                                                  out + SIG_OFF);
}
